// TemporalAttention_89429809037581
// MI455X (gfx1250) — matrix operand order repaired and checked by reading, not compiled, not run on hardware
//
#include <hip/hip_runtime.h>
#include <hip/hip_bf16.h>
#include <math.h>

// ---------------------------------------------------------------------------
// Problem constants (match the JAX reference)
// ---------------------------------------------------------------------------
#define B_ 4
#define T_ 2048
#define D_ 1024
#define H_ 16
#define DK_ 64
#define M_ (B_ * T_)          // 8192 rows of X
#define LN_EPS 1e-5f

// GEMM register blocking: each wave computes (GMT*16) x (GNT*16)
#define GMT 4
#define GNT 2

typedef _Float16 half_t;
typedef __attribute__((ext_vector_type(16))) _Float16 v16h;
typedef __attribute__((ext_vector_type(8)))  _Float16 v8h;
typedef __attribute__((ext_vector_type(8)))  float    v8f;

union V16U { v16h v; v8h h[2]; };

// ---------------------------------------------------------------------------
// Half-wave (16-lane) row reductions for the WMMA f32 C/D layout: a matrix row
// lives in one VGPR across 16 lanes of a half-wave, so xor masks 1,2,4,8
// reduce it without leaving the half.
// ---------------------------------------------------------------------------
__device__ __forceinline__ float hmax16(float v) {
    v = fmaxf(v, __shfl_xor(v, 1, 32));
    v = fmaxf(v, __shfl_xor(v, 2, 32));
    v = fmaxf(v, __shfl_xor(v, 4, 32));
    v = fmaxf(v, __shfl_xor(v, 8, 32));
    return v;
}
__device__ __forceinline__ float hsum16(float v) {
    v += __shfl_xor(v, 1, 32);
    v += __shfl_xor(v, 2, 32);
    v += __shfl_xor(v, 4, 32);
    v += __shfl_xor(v, 8, 32);
    return v;
}

// ---------------------------------------------------------------------------
// f32 -> f16 cast
// ---------------------------------------------------------------------------
__global__ __launch_bounds__(256) void k_cast_f16(const float* __restrict__ src,
                                                  half_t* __restrict__ dst, int n) {
    int i = blockIdx.x * 256 + threadIdx.x;
    if (i < n) dst[i] = (half_t)src[i];
}

// W[k][n] (f32, DxD) -> Wt[n][k] (f16): WMMA B operand contiguous in k
__global__ __launch_bounds__(256) void k_cast_transpose_f16(const float* __restrict__ W,
                                                            half_t* __restrict__ Wt) {
    int i = blockIdx.x * 256 + threadIdx.x;
    if (i < D_ * D_) {
        int k = i / D_, n = i % D_;
        Wt[(size_t)n * D_ + k] = (half_t)W[(size_t)k * D_ + n];
    }
}

// ---------------------------------------------------------------------------
// Register-blocked GEMM: Out = f16( (A[MxK] @ Wt[N][K]^T + bias) * scale )
// One wave computes a 64x32 macro-tile: per K-step of 32 it loads 4 A operands
// + 2 B operands and issues 8 v_wmma_f32_16x16x32_f16 (24 B/WMMA of operand
// traffic vs 64 B/WMMA unblocked).
// store_mode 0: row-major Out[M][N]
// store_mode 1: V-transposed layout Vt[(b*H+h)*DK+dk][T]
// ---------------------------------------------------------------------------
__global__ __launch_bounds__(256)
void k_gemm_f16out(const half_t* __restrict__ A, const half_t* __restrict__ Wt,
                   const float* __restrict__ bias, half_t* __restrict__ Out,
                   int M, int N, int K, float scale, int store_mode) {
    const int lane = threadIdx.x & 31;
    const int lo   = lane & 15;
    const int hi   = lane >> 4;
    const int nblocks = (N >> 4) / GNT;
    const int mblocks = (M >> 4) / GMT;
    const int tid = blockIdx.x * 8 + (threadIdx.x >> 5);
    if (tid >= mblocks * nblocks) return;
    const int nb = tid % nblocks;
    const int mb = tid / nblocks;
    const int m0 = mb * (GMT * 16);
    const int n0 = nb * (GNT * 16);

    const int koffA = hi * 16;
    const int koffB = hi * 16;

    const half_t* arow[GMT];
    const half_t* brow[GNT];
#pragma unroll
    for (int i = 0; i < GMT; ++i) arow[i] = A  + (size_t)(m0 + i * 16 + lo) * K;
#pragma unroll
    for (int j = 0; j < GNT; ++j) brow[j] = Wt + (size_t)(n0 + j * 16 + lo) * K;

    v8f acc[GMT][GNT];
#pragma unroll
    for (int i = 0; i < GMT; ++i)
#pragma unroll
        for (int j = 0; j < GNT; ++j)
            acc[i][j] = (v8f){0.f,0.f,0.f,0.f,0.f,0.f,0.f,0.f};

    for (int kk = 0; kk < K; kk += 32) {
        V16U a[GMT];
        v16h b[GNT];
#pragma unroll
        for (int i = 0; i < GMT; ++i) {
            a[i].h[0] = *(const v8h*)(arow[i] + kk + koffA);
            a[i].h[1] = *(const v8h*)(arow[i] + kk + koffA + 8);
        }
#pragma unroll
        for (int j = 0; j < GNT; ++j)
            b[j] = *(const v16h*)(brow[j] + kk + koffB);
#pragma unroll
        for (int i = 0; i < GMT; ++i)
#pragma unroll
            for (int j = 0; j < GNT; ++j)
                acc[i][j] = __builtin_amdgcn_wmma_f32_16x16x32_f16(
                    false, a[i].v, false, b[j], (short)0, acc[i][j], false, false);
    }

#pragma unroll
    for (int j = 0; j < GNT; ++j) {
        const int col = n0 + j * 16 + lo;
        const float bc = bias[col];
#pragma unroll
        for (int i = 0; i < GMT; ++i) {
#pragma unroll
            for (int r = 0; r < 8; ++r) {
                int row = m0 + i * 16 + r + hi * 8;
                half_t val = (half_t)((acc[i][j][r] + bc) * scale);
                if (store_mode == 0) {
                    Out[(size_t)row * N + col] = val;
                } else {
                    int t  = row % T_;
                    int bb = row / T_;
                    int hh = col >> 6;
                    int dk = col & 63;
                    Out[((size_t)((bb * H_ + hh) * DK_ + dk)) * T_ + t] = val;
                }
            }
        }
    }
}

// Same blocked GEMM with f32 output (Wo projection before residual+LN)
__global__ __launch_bounds__(256)
void k_gemm_f32out(const half_t* __restrict__ A, const half_t* __restrict__ Wt,
                   const float* __restrict__ bias, float* __restrict__ Out,
                   int M, int N, int K) {
    const int lane = threadIdx.x & 31;
    const int lo = lane & 15, hi = lane >> 4;
    const int nblocks = (N >> 4) / GNT;
    const int mblocks = (M >> 4) / GMT;
    const int tid = blockIdx.x * 8 + (threadIdx.x >> 5);
    if (tid >= mblocks * nblocks) return;
    const int nb = tid % nblocks;
    const int mb = tid / nblocks;
    const int m0 = mb * (GMT * 16);
    const int n0 = nb * (GNT * 16);
    const int koffA = hi * 16, koffB = hi * 16;

    const half_t* arow[GMT];
    const half_t* brow[GNT];
#pragma unroll
    for (int i = 0; i < GMT; ++i) arow[i] = A  + (size_t)(m0 + i * 16 + lo) * K;
#pragma unroll
    for (int j = 0; j < GNT; ++j) brow[j] = Wt + (size_t)(n0 + j * 16 + lo) * K;

    v8f acc[GMT][GNT];
#pragma unroll
    for (int i = 0; i < GMT; ++i)
#pragma unroll
        for (int j = 0; j < GNT; ++j)
            acc[i][j] = (v8f){0.f,0.f,0.f,0.f,0.f,0.f,0.f,0.f};

    for (int kk = 0; kk < K; kk += 32) {
        V16U a[GMT];
        v16h b[GNT];
#pragma unroll
        for (int i = 0; i < GMT; ++i) {
            a[i].h[0] = *(const v8h*)(arow[i] + kk + koffA);
            a[i].h[1] = *(const v8h*)(arow[i] + kk + koffA + 8);
        }
#pragma unroll
        for (int j = 0; j < GNT; ++j)
            b[j] = *(const v16h*)(brow[j] + kk + koffB);
#pragma unroll
        for (int i = 0; i < GMT; ++i)
#pragma unroll
            for (int j = 0; j < GNT; ++j)
                acc[i][j] = __builtin_amdgcn_wmma_f32_16x16x32_f16(
                    false, a[i].v, false, b[j], (short)0, acc[i][j], false, false);
    }

#pragma unroll
    for (int j = 0; j < GNT; ++j) {
        const int col = n0 + j * 16 + lo;
        const float bc = bias[col];
#pragma unroll
        for (int i = 0; i < GMT; ++i)
#pragma unroll
            for (int r = 0; r < 8; ++r) {
                int row = m0 + i * 16 + r + hi * 8;
                Out[(size_t)row * N + col] = acc[i][j][r] + bc;
            }
    }
}

// ---------------------------------------------------------------------------
// Flash attention (causal). One wave = 32 queries (two 16-row tiles) of one
// head, so every K-tile / V-tile B operand load is reused by two WMMAs.
// Qh: [B*T, D] f16 pre-scaled by 1/sqrt(DK). Kh: [B*T, D] f16.
// Vt: [(b*H+h)*DK+dk][T] f16. Ctxh out: [B*T, D] f16.
// Per 32-key block: 8 score WMMAs + 8 context WMMAs on 8 global B loads;
// P tiles round-trip through LDS (C-layout -> A-layout).
// ---------------------------------------------------------------------------
__global__ __launch_bounds__(256)
void k_attn(const half_t* __restrict__ Qh, const half_t* __restrict__ Kh,
            const half_t* __restrict__ Vt, half_t* __restrict__ Ctxh) {
    __shared__ _Float16 plds[8][2][16][32];   // 16 KB: two 16x32 P tiles per wave

    const int w    = threadIdx.x >> 5;
    const int lane = threadIdx.x & 31;
    const int lo   = lane & 15;
    const int hi   = lane >> 4;

    const int gw = blockIdx.x * 8 + w;
    const int QB = T_ / 32;
    const int qb   = gw % QB;
    const int head = (gw / QB) % H_;
    const int bb   =  gw / (QB * H_);

    const int q0   = qb * 32;
    const int hcol = head * DK_;
    const int koffA = hi * 16;
    const int koffB = hi * 16;

    // Q tiles, A-layout, both K=32 slices (dk 0-31 and 32-63)
    V16U qa[2][2];
#pragma unroll
    for (int u = 0; u < 2; ++u) {
        const half_t* qrow = Qh + (size_t)(bb * T_ + q0 + u * 16 + lo) * D_ + hcol;
        qa[u][0].h[0] = *(const v8h*)(qrow + koffA);
        qa[u][0].h[1] = *(const v8h*)(qrow + koffA + 8);
        qa[u][1].h[0] = *(const v8h*)(qrow + 32 + koffA);
        qa[u][1].h[1] = *(const v8h*)(qrow + 32 + koffA + 8);
    }

    v8f acc[2][4];
    float mst[2][8], lst[2][8];
#pragma unroll
    for (int u = 0; u < 2; ++u) {
#pragma unroll
        for (int t = 0; t < 4; ++t) acc[u][t] = (v8f){0.f,0.f,0.f,0.f,0.f,0.f,0.f,0.f};
#pragma unroll
        for (int r = 0; r < 8; ++r) { mst[u][r] = -INFINITY; lst[u][r] = 0.f; }
    }

    const int qend = q0 + 31;
    const half_t* vbase = Vt + (size_t)((bb * H_ + head) * DK_) * T_;

    for (int kb = 0; kb <= qend; kb += 32) {
        // ---- K B-operands: shared by both query tiles ---------------------
        const half_t* k0 = Kh + (size_t)(bb * T_ + kb + lo) * D_ + hcol + koffB;
        const half_t* k1 = Kh + (size_t)(bb * T_ + kb + 16 + lo) * D_ + hcol + koffB;
        v16h b00 = *(const v16h*)(k0);
        v16h b01 = *(const v16h*)(k0 + 32);
        v16h b10 = *(const v16h*)(k1);
        v16h b11 = *(const v16h*)(k1 + 32);

        const int key0 = kb + lo;
        const int key1 = kb + 16 + lo;

#pragma unroll
        for (int u = 0; u < 2; ++u) {
            v8f s0 = {0.f,0.f,0.f,0.f,0.f,0.f,0.f,0.f};
            v8f s1 = {0.f,0.f,0.f,0.f,0.f,0.f,0.f,0.f};
            s0 = __builtin_amdgcn_wmma_f32_16x16x32_f16(false, qa[u][0].v, false, b00,
                                                        (short)0, s0, false, false);
            s0 = __builtin_amdgcn_wmma_f32_16x16x32_f16(false, qa[u][1].v, false, b01,
                                                        (short)0, s0, false, false);
            s1 = __builtin_amdgcn_wmma_f32_16x16x32_f16(false, qa[u][0].v, false, b10,
                                                        (short)0, s1, false, false);
            s1 = __builtin_amdgcn_wmma_f32_16x16x32_f16(false, qa[u][1].v, false, b11,
                                                        (short)0, s1, false, false);

            // ---- causal mask + online softmax -----------------------------
#pragma unroll
            for (int r = 0; r < 8; ++r) {
                const int q = q0 + u * 16 + r + hi * 8;
                float v0 = (key0 <= q) ? s0[r] : -3.0e38f;
                float v1 = (key1 <= q) ? s1[r] : -3.0e38f;
                float rowm = hmax16(fmaxf(v0, v1));
                float nm = fmaxf(mst[u][r], rowm);
                float sc = __expf(mst[u][r] - nm);   // exp(-inf) = 0 first block
                float p0 = __expf(v0 - nm);
                float p1 = __expf(v1 - nm);
                lst[u][r] = lst[u][r] * sc + hsum16(p0 + p1);
                mst[u][r] = nm;
#pragma unroll
                for (int t = 0; t < 4; ++t) acc[u][t][r] *= sc;
                plds[w][u][r + hi * 8][lo]      = (_Float16)p0;
                plds[w][u][r + hi * 8][16 + lo] = (_Float16)p1;
            }
        }
        asm volatile("s_wait_dscnt 0" ::: "memory");

        // ---- reload P in A layout, ctx += P @ V (V loads shared) ----------
        V16U pa[2];
#pragma unroll
        for (int u = 0; u < 2; ++u) {
            pa[u].h[0] = *(const v8h*)&plds[w][u][lo][koffA];
            pa[u].h[1] = *(const v8h*)&plds[w][u][lo][koffA + 8];
        }
#pragma unroll
        for (int t = 0; t < 4; ++t) {
            v16h vb = *(const v16h*)(vbase + (size_t)(t * 16 + lo) * T_ + kb + koffB);
            acc[0][t] = __builtin_amdgcn_wmma_f32_16x16x32_f16(false, pa[0].v, false, vb,
                                                               (short)0, acc[0][t], false, false);
            acc[1][t] = __builtin_amdgcn_wmma_f32_16x16x32_f16(false, pa[1].v, false, vb,
                                                               (short)0, acc[1][t], false, false);
        }
    }

    // ---- normalize and store context (f16, [B*T, D]) ----------------------
#pragma unroll
    for (int u = 0; u < 2; ++u) {
#pragma unroll
        for (int r = 0; r < 8; ++r) {
            float inv = 1.0f / lst[u][r];
            int row = bb * T_ + q0 + u * 16 + r + hi * 8;
#pragma unroll
            for (int t = 0; t < 4; ++t) {
                Ctxh[(size_t)row * D_ + hcol + t * 16 + lo] = (half_t)(acc[u][t][r] * inv);
            }
        }
    }
}

// ---------------------------------------------------------------------------
// Residual + LayerNorm: out = LN(proj + x) * g + b, one row per block
// ---------------------------------------------------------------------------
__global__ __launch_bounds__(256)
void k_ln(const float* __restrict__ proj, const float* __restrict__ x,
          const float* __restrict__ g, const float* __restrict__ beta,
          float* __restrict__ out) {
    __shared__ float shs[8], shs2[8];
    const int row = blockIdx.x;
    const int tid = threadIdx.x;
    const size_t base = (size_t)row * D_;

    float vals[4];
    float s = 0.f, s2 = 0.f;
#pragma unroll
    for (int i = 0; i < 4; ++i) {
        int idx = tid + i * 256;
        float v = proj[base + idx] + x[base + idx];
        vals[i] = v;
        s += v; s2 += v * v;
    }
#pragma unroll
    for (int m = 1; m < 32; m <<= 1) {
        s  += __shfl_xor(s,  m, 32);
        s2 += __shfl_xor(s2, m, 32);
    }
    if ((tid & 31) == 0) { shs[tid >> 5] = s; shs2[tid >> 5] = s2; }
    __syncthreads();
    float ts = 0.f, ts2 = 0.f;
#pragma unroll
    for (int i = 0; i < 8; ++i) { ts += shs[i]; ts2 += shs2[i]; }
    const float mu  = ts * (1.0f / D_);
    const float var = ts2 * (1.0f / D_) - mu * mu;
    const float rstd = rsqrtf(var + LN_EPS);
#pragma unroll
    for (int i = 0; i < 4; ++i) {
        int idx = tid + i * 256;
        out[base + idx] = (vals[i] - mu) * rstd * g[idx] + beta[idx];
    }
}

// ---------------------------------------------------------------------------
// Host-side orchestration
// ---------------------------------------------------------------------------
extern "C" void kernel_launch(void* const* d_in, const int* in_sizes, int n_in,
                              void* d_out, int out_size, void* d_ws, size_t ws_size,
                              hipStream_t stream) {
    (void)in_sizes; (void)n_in; (void)out_size; (void)ws_size;
    const float* x    = (const float*)d_in[0];
    const float* Wq   = (const float*)d_in[1];
    const float* bq   = (const float*)d_in[2];
    const float* Wk   = (const float*)d_in[3];
    const float* bk   = (const float*)d_in[4];
    const float* Wv   = (const float*)d_in[5];
    const float* bv   = (const float*)d_in[6];
    const float* Wo   = (const float*)d_in[7];
    const float* bo   = (const float*)d_in[8];
    const float* ln_g = (const float*)d_in[9];
    const float* ln_b = (const float*)d_in[10];
    float* out = (float*)d_out;

    // Workspace layout (bytes)
    char* ws = (char*)d_ws;
    const size_t MB = (size_t)1 << 20;
    half_t* Xh   = (half_t*)(ws + 0 * MB);    // 16 MB  (8192x1024 f16)
    half_t* Wq_t = (half_t*)(ws + 16 * MB);   //  2 MB
    half_t* Wk_t = (half_t*)(ws + 18 * MB);   //  2 MB
    half_t* Wv_t = (half_t*)(ws + 20 * MB);   //  2 MB
    half_t* Wo_t = (half_t*)(ws + 22 * MB);   //  2 MB
    half_t* Qh   = (half_t*)(ws + 24 * MB);   // 16 MB
    half_t* Kh   = (half_t*)(ws + 40 * MB);   // 16 MB
    half_t* Vt   = (half_t*)(ws + 56 * MB);   // 16 MB (transposed per head)
    half_t* Ctxh = (half_t*)(ws + 72 * MB);   // 16 MB
    float*  Proj = (float*)(ws + 88 * MB);    // 32 MB  -> 120 MB total

    const int nX = M_ * D_;
    k_cast_f16<<<(nX + 255) / 256, 256, 0, stream>>>(x, Xh, nX);
    const int nW = D_ * D_;
    k_cast_transpose_f16<<<(nW + 255) / 256, 256, 0, stream>>>(Wq, Wq_t);
    k_cast_transpose_f16<<<(nW + 255) / 256, 256, 0, stream>>>(Wk, Wk_t);
    k_cast_transpose_f16<<<(nW + 255) / 256, 256, 0, stream>>>(Wv, Wv_t);
    k_cast_transpose_f16<<<(nW + 255) / 256, 256, 0, stream>>>(Wo, Wo_t);

    // 64x32 macro-tiles: (8192/64) * (1024/32) = 128*32 = 4096 waves
    const int macroTiles = (M_ / (GMT * 16)) * (D_ / (GNT * 16));
    const int gblocks = (macroTiles + 7) / 8;       // 512
    const float qscale = 0.125f;                    // 1/sqrt(DK)
    k_gemm_f16out<<<gblocks, 256, 0, stream>>>(Xh, Wq_t, bq, Qh, M_, D_, D_, qscale, 0);
    k_gemm_f16out<<<gblocks, 256, 0, stream>>>(Xh, Wk_t, bk, Kh, M_, D_, D_, 1.0f, 0);
    k_gemm_f16out<<<gblocks, 256, 0, stream>>>(Xh, Wv_t, bv, Vt, M_, D_, D_, 1.0f, 1);

    const int attnWaves = B_ * H_ * (T_ / 32);      // 4096 (32 queries per wave)
    k_attn<<<attnWaves / 8, 256, 0, stream>>>(Qh, Kh, Vt, Ctxh);

    k_gemm_f32out<<<gblocks, 256, 0, stream>>>(Ctxh, Wo_t, bo, Proj, M_, D_, D_);
    k_ln<<<M_, 256, 0, stream>>>(Proj, x, ln_g, ln_b, out);
}
